// GraphNeuralNetwork_14396730377018
// MI455X (gfx1250) — matrix operand order repaired and checked by reading, not compiled, not run on hardware
//
#include <hip/hip_runtime.h>

#define N_NODES 20000
#define N_EDGES 240000
#define NGRAPH  64
#define FDIM    128
#define FEDIM   64
#define HEADS   4
#define CDIM    128
#define HC      512
#define LN_EPS  1e-5f
#define AVGLOG  2.5649493574615367f

typedef __attribute__((ext_vector_type(16))) _Float16 v16h;
typedef __attribute__((ext_vector_type(8)))  _Float16 v8h;
typedef __attribute__((ext_vector_type(8)))  float    v8f;

// ---------- monotonic float <-> uint encoding for atomic min/max ----------
__device__ __forceinline__ unsigned fenc(float f) {
  unsigned u = __float_as_uint(f);
  return (u & 0x80000000u) ? ~u : (u | 0x80000000u);
}
__device__ __forceinline__ float fdec(unsigned u) {
  return __uint_as_float((u & 0x80000000u) ? (u ^ 0x80000000u) : ~u);
}

// =====================================================================
// Generic WMMA GEMM:  out[M,Nout] = act( A[M,K] @ W[K,Nout] + bias )
// A = up to 3 concatenated row-gathered segments (fused gather-GEMM).
// Tile: 32 rows x 64 cols per block (4 waves; each wave does 2 WMMAs
// per K-step sharing one B fragment). LDS tiles are stored directly in
// CDNA5 WMMA fragment order so fragment fetch = 2x ds_load_b128.
//   A frag (16-bit, 16x32): lane<16 holds K{0-15},      lane>=16 K{16-31}
//   B frag (16-bit, 32x16): lane<16 holds K{0-15},      lane>=16 K{16-31}
// =====================================================================
__global__ void __launch_bounds__(128)
gemm_wmma(const float* __restrict__ a0, const int* __restrict__ i0, int w0,
          const float* __restrict__ a1, const int* __restrict__ i1, int w1,
          const float* __restrict__ a2, const int* __restrict__ i2, int w2,
          int K, const float* __restrict__ W, const float* __restrict__ bias,
          float* __restrict__ out, int M, int Nout, int act)
{
  __shared__ __attribute__((aligned(32))) _Float16 aF[2][32][16]; // [sub][lane][elem]
  __shared__ __attribute__((aligned(32))) _Float16 bF[4][32][16]; // [wave][lane][elem]
  const int tid  = threadIdx.x;
  const int wave = tid >> 5;
  const int lane = tid & 31;
  const int row0 = blockIdx.x << 5;
  const int col0 = blockIdx.y << 6;

  // ---- this thread's A staging slot: one 8-half group ----
  const int sA   = tid >> 6;          // subtile 0/1
  const int lnA  = (tid >> 1) & 31;   // target fragment lane
  const int hfA  = tid & 1;           // which 8-elem half of the fragment
  const int kloc = ((lnA >> 4) << 4) + (hfA << 3);   // 0 / 8 / 16 / 24
  const int arow = row0 + (sA << 4) + (lnA & 15);
  // hoist gather indices / segment bases out of the K loop
  int r0 = 0, r1 = 0, r2 = 0;
  if (arow < M) {
    r0 = i0 ? i0[arow] : arow;
    r1 = i1 ? i1[arow] : arow;
    r2 = i2 ? i2[arow] : arow;
  }
  const float* base0 = a0 + (size_t)r0 * w0;
  const float* base1 = a1 ? a1 + (size_t)r1 * w1 : nullptr;
  const float* base2 = a2 ? a2 + (size_t)r2 * w2 : nullptr;
  const bool fullM = (row0 + 32 <= M);
  const bool fullN = (col0 + 64 <= Nout);

  v8f acc0 = {}, acc1 = {};

  for (int kk = 0; kk < K; kk += 32) {
    const bool fullK = (kk + 32 <= K);
    // ---------- stage A (one 8-float group per thread) ----------
    {
      _Float16 hv[8];
      int gk = kk + kloc;
      if (fullK && fullM) {           // interior: segment widths are 8-aligned
        const float* p;
        int k = gk;
        if (k < w0) p = base0 + k;
        else { k -= w0; if (k < w1) p = base1 + k; else p = base2 + (k - w1); }
#pragma unroll
        for (int j = 0; j < 8; ++j) hv[j] = (_Float16)p[j];
      } else {
#pragma unroll
        for (int j = 0; j < 8; ++j) {
          float v = 0.0f;
          int k = gk + j;
          if (arow < M && k < K) {
            if (k < w0) v = base0[k];
            else { k -= w0; v = (k < w1) ? base1[k] : base2[k - w1]; }
          }
          hv[j] = (_Float16)v;
        }
      }
      v8h pk;
#pragma unroll
      for (int j = 0; j < 8; ++j) pk[j] = hv[j];
      *(v8h*)&aF[sA][lnA][hfA << 3] = pk;   // one 16B LDS store
    }
    // ---------- stage B (two 8-col groups per thread) ----------
#pragma unroll
    for (int t = 0; t < 2; ++t) {
      int gidx = tid + (t << 7);            // 0..255
      int r  = gidx >> 3;                   // K row 0..31
      int c8 = (gidx & 7) << 3;             // col start
      int gk = kk + r;
      float vv[8];
      if (fullK && fullN) {
        const float* p = W + (size_t)gk * Nout + (col0 + c8);
#pragma unroll
        for (int j = 0; j < 8; ++j) vv[j] = p[j];
      } else {
#pragma unroll
        for (int j = 0; j < 8; ++j) {
          int gc = col0 + c8 + j;
          vv[j] = (gk < K && gc < Nout) ? W[(size_t)gk * Nout + gc] : 0.0f;
        }
      }
      int hi = (r >> 4) << 4;
      int idx = r & 15;
#pragma unroll
      for (int j = 0; j < 8; ++j) {
        int c = c8 + j;
        bF[c >> 4][(c & 15) + hi][idx] = (_Float16)vv[j];
      }
    }
    __syncthreads();
    // ---------- fragments: contiguous 32B per lane ----------
    v16h af0 = *(const v16h*)&aF[0][lane][0];
    v16h af1 = *(const v16h*)&aF[1][lane][0];
    v16h bfv = *(const v16h*)&bF[wave][lane][0];
    acc0 = __builtin_amdgcn_wmma_f32_16x16x32_f16(false, af0, false, bfv,
                                                  (short)0, acc0, false, false);
    acc1 = __builtin_amdgcn_wmma_f32_16x16x32_f16(false, af1, false, bfv,
                                                  (short)0, acc1, false, false);
    __syncthreads();
  }

  // ---- epilogue: C/D layout lane<16: N=lane, M=j ; lane>=16: M=8+j ----
  int ccol  = col0 + (wave << 4) + (lane & 15);
  int mbase = (lane >> 4) << 3;
  if (ccol < Nout) {
    float bv = bias ? bias[ccol] : 0.0f;
#pragma unroll
    for (int j = 0; j < 8; ++j) {
      int grow = row0 + mbase + j;
      if (grow < M) {
        float v = acc0[j] + bv;
        if (act) v = fmaxf(v, 0.0f);
        out[(size_t)grow * Nout + ccol] = v;
      }
      int grow1 = grow + 16;
      if (grow1 < M) {
        float v = acc1[j] + bv;
        if (act) v = fmaxf(v, 0.0f);
        out[(size_t)grow1 * Nout + ccol] = v;
      }
    }
  }
}

// =====================================================================
// LayerNorm: wave per row (width 64 or 128)
// =====================================================================
__global__ void __launch_bounds__(256)
layernorm_k(const float* __restrict__ in, float* __restrict__ out,
            const float* __restrict__ g, const float* __restrict__ b,
            int rows, int width)
{
  int w = threadIdx.x >> 5, lane = threadIdx.x & 31;
  int row = blockIdx.x * 8 + w;
  if (row >= rows) return;
  const float* p = in + (size_t)row * width;
  int nv = width >> 5;
  float vals[4];
  float s = 0.f;
  for (int i = 0; i < nv; ++i) { vals[i] = p[lane + (i << 5)]; s += vals[i]; }
  for (int m = 16; m; m >>= 1) s += __shfl_xor(s, m, 32);
  float mean = s / (float)width;
  float vv = 0.f;
  for (int i = 0; i < nv; ++i) { float d = vals[i] - mean; vv += d * d; }
  for (int m = 16; m; m >>= 1) vv += __shfl_xor(vv, m, 32);
  float inv = rsqrtf(vv / (float)width + LN_EPS);
  float* o = out + (size_t)row * width;
  for (int i = 0; i < nv; ++i) {
    int c = lane + (i << 5);
    o[c] = g[c] * (vals[i] - mean) * inv + b[c];
  }
}

// =====================================================================
// GraphNorm pieces
// =====================================================================
__global__ void gn_accum1(const float* __restrict__ x, const int* __restrict__ batch,
                          float* __restrict__ macc, float* __restrict__ gcnt, int n)
{
  int t = blockIdx.x * 256 + threadIdx.x;
  if (t >= n * FDIM) return;
  int node = t >> 7, f = t & 127;
  int g = batch[node];
  atomicAdd(&macc[g * FDIM + f], x[t]);
  if (f == 0) atomicAdd(&gcnt[g], 1.0f);
}
__global__ void gn_div(float* __restrict__ macc, const float* __restrict__ gcnt)
{
  int t = blockIdx.x * 256 + threadIdx.x;
  if (t >= NGRAPH * FDIM) return;
  macc[t] /= fmaxf(gcnt[t >> 7], 1.0f);
}
__global__ void gn_accum2(const float* __restrict__ x, const int* __restrict__ batch,
                          const float* __restrict__ macc, const float* __restrict__ ms,
                          float* __restrict__ sh, float* __restrict__ vacc, int n)
{
  int t = blockIdx.x * 256 + threadIdx.x;
  if (t >= n * FDIM) return;
  int node = t >> 7, f = t & 127;
  int g = batch[node];
  float s = x[t] - ms[f] * macc[g * FDIM + f];
  sh[t] = s;
  atomicAdd(&vacc[g * FDIM + f], s * s);
}
__global__ void gn_finv(float* __restrict__ vacc, const float* __restrict__ gcnt)
{
  int t = blockIdx.x * 256 + threadIdx.x;
  if (t >= NGRAPH * FDIM) return;
  vacc[t] = rsqrtf(vacc[t] / fmaxf(gcnt[t >> 7], 1.0f) + LN_EPS);
}
__global__ void gn_apply(float* __restrict__ sh, const int* __restrict__ batch,
                         const float* __restrict__ vacc, const float* __restrict__ w,
                         const float* __restrict__ b, int n)
{
  int t = blockIdx.x * 256 + threadIdx.x;
  if (t >= n * FDIM) return;
  int node = t >> 7, f = t & 127;
  int g = batch[node];
  sh[t] = w[f] * sh[t] * vacc[g * FDIM + f] + b[f];
}

// =====================================================================
// PNA aggregation
// =====================================================================
__global__ void cnt_k(const int* __restrict__ dst, float* __restrict__ cnt, int E)
{
  int e = blockIdx.x * 256 + threadIdx.x;
  if (e < E) atomicAdd(&cnt[dst[e]], 1.0f);
}
__global__ void pna_agg(const float* __restrict__ m, const int* __restrict__ dst,
                        float* __restrict__ sum, float* __restrict__ sum2,
                        unsigned* __restrict__ mnb, unsigned* __restrict__ mxb, int E)
{
  int t = blockIdx.x * 256 + threadIdx.x;
  if (t >= E * FDIM) return;
  int e = t >> 7, f = t & 127;
  int d = dst[e];
  float v = m[t];
  size_t o = (size_t)d * FDIM + f;
  atomicAdd(&sum[o], v);
  atomicAdd(&sum2[o], v * v);
  unsigned en = fenc(v);
  atomicMin(&mnb[o], en);
  atomicMax(&mxb[o], en);
}
__global__ void pna_build(const float* __restrict__ hgn, const float* __restrict__ sum,
                          const float* __restrict__ sum2, const unsigned* __restrict__ mnb,
                          const unsigned* __restrict__ mxb, const float* __restrict__ cnt,
                          float* __restrict__ A, int n)
{
  int t = blockIdx.x * 256 + threadIdx.x;
  if (t >= n * FDIM) return;
  int node = t >> 7, f = t & 127;
  float c = cnt[node], cc = fmaxf(c, 1.0f);
  float mean = sum[t] / cc, m2 = sum2[t] / cc;
  float sd = sqrtf(fmaxf(m2 - mean * mean, 0.0f) + LN_EPS);
  float vmn = (c > 0.0f) ? fdec(mnb[t]) : 0.0f;
  float vmx = (c > 0.0f) ? fdec(mxb[t]) : 0.0f;
  float logd = logf(cc + 1.0f);
  float s1 = logd / AVGLOG, s2 = AVGLOG / logd;
  float* row = A + (size_t)node * (13 * FDIM);
  row[f]          = hgn[t];
  row[128 + f]    = mean;      row[256 + f]  = vmn;      row[384 + f]  = vmx;      row[512 + f]  = sd;
  row[640 + f]    = mean * s1; row[768 + f]  = vmn * s1; row[896 + f]  = vmx * s1; row[1024 + f] = sd * s1;
  row[1152 + f]   = mean * s2; row[1280 + f] = vmn * s2; row[1408 + f] = vmx * s2; row[1536 + f] = sd * s2;
}
__global__ void add_k(float* __restrict__ a, const float* __restrict__ b, int n)
{
  int t = blockIdx.x * 256 + threadIdx.x;
  if (t < n) a[t] += b[t];
}

// =====================================================================
// Transformer attention
// =====================================================================
__global__ void __launch_bounds__(256)
attn_alpha(const float* __restrict__ q, const float* __restrict__ k,
           const float* __restrict__ ee, const int* __restrict__ src,
           const int* __restrict__ dst, float* __restrict__ alpha,
           unsigned* __restrict__ amax, int E)
{
  int wid = blockIdx.x * 8 + (threadIdx.x >> 5);
  int lane = threadIdx.x & 31;
  if (wid >= E * HEADS) return;
  int e = wid >> 2, hh = wid & 3;
  int sN = src[e], dN = dst[e];
  const float* qp = q + (size_t)dN * HC + hh * CDIM;
  const float* kp = k + (size_t)sN * HC + hh * CDIM;
  const float* ep = ee + (size_t)e * HC + hh * CDIM;
  float sum = 0.f;
#pragma unroll
  for (int i = 0; i < 4; ++i) {
    int c = lane + (i << 5);
    sum += qp[c] * (kp[c] + ep[c]);
  }
  for (int m = 16; m; m >>= 1) sum += __shfl_xor(sum, m, 32);
  sum *= 0.0883883476483184f;   // 1/sqrt(128)
  if (lane == 0) {
    alpha[wid] = sum;
    atomicMax(&amax[(size_t)dN * HEADS + hh], fenc(sum));
  }
}
__global__ void amax_fin(unsigned* __restrict__ amax, int n)
{
  int t = blockIdx.x * 256 + threadIdx.x;
  if (t >= n) return;
  unsigned u = amax[t];
  ((float*)amax)[t] = (u == 0u) ? 0.0f : fdec(u);   // untouched (no in-edges) -> 0
}
__global__ void attn_ea(float* __restrict__ alpha, const float* __restrict__ amaxf,
                        const int* __restrict__ dst, float* __restrict__ asum, int EH)
{
  int t = blockIdx.x * 256 + threadIdx.x;
  if (t >= EH) return;
  int e = t >> 2, hh = t & 3;
  int d = dst[e];
  float ea = expf(alpha[t] - amaxf[d * HEADS + hh]);
  alpha[t] = ea;
  atomicAdd(&asum[d * HEADS + hh], ea);
}
__global__ void attn_out(const float* __restrict__ vmat, const float* __restrict__ eet,
                         const float* __restrict__ alpha, const float* __restrict__ asum,
                         const int* __restrict__ src, const int* __restrict__ dst,
                         float* __restrict__ attout, int total)
{
  int t = blockIdx.x * 256 + threadIdx.x;
  if (t >= total) return;
  int e = t >> 9, c = t & 511;
  int hh = c >> 7;
  int s = src[e], d = dst[e];
  float w = alpha[(size_t)e * HEADS + hh] / asum[(size_t)d * HEADS + hh];
  float val = (vmat[(size_t)s * HC + c] + eet[(size_t)e * HC + c]) * w;
  atomicAdd(&attout[(size_t)d * HC + c], val);
}
__global__ void __launch_bounds__(256)
trans_combine(float* __restrict__ h, const float* __restrict__ attout,
              const float* __restrict__ xr, const float* __restrict__ bw, int n)
{
  int w = threadIdx.x >> 5, lane = threadIdx.x & 31;
  int node = blockIdx.x * 8 + w;
  if (node >= n) return;
  const float* ao = attout + (size_t)node * HC;
  const float* xp = xr + (size_t)node * FDIM;
  float om[4], xv[4], partial = 0.f;
#pragma unroll
  for (int i = 0; i < 4; ++i) {
    int f = lane + (i << 5);
    om[i] = 0.25f * (ao[f] + ao[128 + f] + ao[256 + f] + ao[384 + f]);
    xv[i] = xp[f];
    partial += om[i] * bw[f] + xv[i] * bw[128 + f] + (om[i] - xv[i]) * bw[256 + f];
  }
  for (int m = 16; m; m >>= 1) partial += __shfl_xor(partial, m, 32);
  float beta = 1.0f / (1.0f + expf(-partial));
  float* hp = h + (size_t)node * FDIM;
#pragma unroll
  for (int i = 0; i < 4; ++i) {
    int f = lane + (i << 5);
    hp[f] += beta * xv[i] + (1.0f - beta) * om[i];
  }
}

// =====================================================================
// Host side
// =====================================================================
static inline int cdiv(long a, long b) { return (int)((a + b - 1) / b); }

struct LinP { const float* w; const float* b; };
struct GnP  { const float* w; const float* b; const float* ms; };
struct PnaP { GnP gn; LinP enc, pre, post, lin; };
struct TrP  { GnP gn; LinP q, k, v, skip; const float* ew; const float* betaw; };
struct AllP {
  LinP emb, edge1, edge2, head1, head2;
  const float *emb_ln_g, *emb_ln_b, *edge_ln_g, *edge_ln_b, *head_ln_g, *head_ln_b;
  PnaP pna1, pna2; TrP tr1, tr2;
};

static void fill_sorted(const float* const* p, AllP& P) {
  P.edge1 = { p[1], p[0] };  P.edge2 = { p[3], p[2] };
  P.edge_ln_b = p[4]; P.edge_ln_g = p[5];
  P.emb = { p[7], p[6] };    P.emb_ln_b = p[8]; P.emb_ln_g = p[9];
  P.head1 = { p[11], p[10] }; P.head2 = { p[13], p[12] };
  P.head_ln_b = p[14]; P.head_ln_g = p[15];
  auto pna = [&](int b, PnaP& q) {
    q.enc  = { p[b + 1], p[b + 0] };
    q.gn   = { p[b + 4], p[b + 2], p[b + 3] };     // weight, bias, mean_scale
    q.lin  = { p[b + 6], p[b + 5] };
    q.post = { p[b + 8], p[b + 7] };
    q.pre  = { p[b + 10], p[b + 9] };
  };
  pna(16, P.pna1); pna(27, P.pna2);
  auto tr = [&](int b, TrP& q) {
    q.betaw = p[b + 0]; q.ew = p[b + 1];
    q.gn   = { p[b + 4], p[b + 2], p[b + 3] };
    q.k    = { p[b + 6], p[b + 5] };
    q.q    = { p[b + 8], p[b + 7] };
    q.skip = { p[b + 10], p[b + 9] };
    q.v    = { p[b + 12], p[b + 11] };
  };
  tr(38, P.tr1); tr(51, P.tr2);
}
static void fill_insertion(const float* const* p, AllP& P) {
  P.emb = { p[0], p[1] };   P.emb_ln_g = p[2]; P.emb_ln_b = p[3];
  P.edge1 = { p[4], p[5] }; P.edge2 = { p[6], p[7] };
  P.edge_ln_g = p[8]; P.edge_ln_b = p[9];
  auto pna = [&](int b, PnaP& q) {
    q.gn = { p[b + 0], p[b + 1], p[b + 2] };
    q.enc = { p[b + 3], p[b + 4] };  q.pre = { p[b + 5], p[b + 6] };
    q.post = { p[b + 7], p[b + 8] }; q.lin = { p[b + 9], p[b + 10] };
  };
  auto tr = [&](int b, TrP& q) {
    q.gn = { p[b + 0], p[b + 1], p[b + 2] };
    q.q = { p[b + 3], p[b + 4] }; q.k = { p[b + 5], p[b + 6] };
    q.v = { p[b + 7], p[b + 8] }; q.ew = p[b + 9];
    q.skip = { p[b + 10], p[b + 11] }; q.betaw = p[b + 12];
  };
  pna(10, P.pna1); tr(21, P.tr1); pna(34, P.pna2); tr(45, P.tr2);
  P.head_ln_g = p[58]; P.head_ln_b = p[59];
  P.head1 = { p[60], p[61] }; P.head2 = { p[62], p[63] };
}

static void gemm1(hipStream_t s, const float* A, int K, const float* W, const float* b,
                  float* out, int M, int Nout, int act) {
  dim3 g(cdiv(M, 32), cdiv(Nout, 64));
  gemm_wmma<<<g, 128, 0, s>>>(A, nullptr, K, nullptr, nullptr, 0, nullptr, nullptr, 0,
                              K, W, b, out, M, Nout, act);
}

static void run_graph_norm(hipStream_t s, const float* x, const int* batch, const GnP& g,
                           float* outx, float* macc, float* vacc, float* gcnt) {
  hipMemsetAsync(macc, 0, NGRAPH * FDIM * sizeof(float), s);
  hipMemsetAsync(vacc, 0, NGRAPH * FDIM * sizeof(float), s);
  hipMemsetAsync(gcnt, 0, NGRAPH * sizeof(float), s);
  int n = N_NODES * FDIM;
  gn_accum1<<<cdiv(n, 256), 256, 0, s>>>(x, batch, macc, gcnt, N_NODES);
  gn_div<<<cdiv(NGRAPH * FDIM, 256), 256, 0, s>>>(macc, gcnt);
  gn_accum2<<<cdiv(n, 256), 256, 0, s>>>(x, batch, macc, g.ms, outx, vacc, N_NODES);
  gn_finv<<<cdiv(NGRAPH * FDIM, 256), 256, 0, s>>>(vacc, gcnt);
  gn_apply<<<cdiv(n, 256), 256, 0, s>>>(outx, batch, vacc, g.w, g.b, N_NODES);
}

static void run_pna(hipStream_t s, const PnaP& P, float* h, float* hgn, const int* batch,
                    const int* src, const int* dst, const float* efeat,
                    float* sum, float* sum2, unsigned* mnb, unsigned* mxb, float* cnt,
                    float* eeenc, float* medge, float* apost, float* t1, float* t2,
                    float* macc, float* vacc, float* gcnt) {
  run_graph_norm(s, h, batch, P.gn, hgn, macc, vacc, gcnt);
  gemm1(s, efeat, FEDIM, P.enc.w, P.enc.b, eeenc, N_EDGES, FDIM, 0);
  { // fused gather GEMM: concat(h[dst], h[src], edge_enc) @ W_pre
    dim3 g(cdiv(N_EDGES, 32), cdiv(FDIM, 64));
    gemm_wmma<<<g, 128, 0, s>>>(hgn, dst, FDIM, hgn, src, FDIM, eeenc, nullptr, FDIM,
                                3 * FDIM, P.pre.w, P.pre.b, medge, N_EDGES, FDIM, 0);
  }
  hipMemsetAsync(sum,  0,    (size_t)N_NODES * FDIM * 4, s);
  hipMemsetAsync(sum2, 0,    (size_t)N_NODES * FDIM * 4, s);
  hipMemsetAsync(mnb,  0xFF, (size_t)N_NODES * FDIM * 4, s);
  hipMemsetAsync(mxb,  0,    (size_t)N_NODES * FDIM * 4, s);
  hipMemsetAsync(cnt,  0,    (size_t)N_NODES * 4, s);
  cnt_k<<<cdiv(N_EDGES, 256), 256, 0, s>>>(dst, cnt, N_EDGES);
  pna_agg<<<cdiv((long)N_EDGES * FDIM, 256), 256, 0, s>>>(medge, dst, sum, sum2, mnb, mxb, N_EDGES);
  pna_build<<<cdiv(N_NODES * FDIM, 256), 256, 0, s>>>(hgn, sum, sum2, mnb, mxb, cnt, apost, N_NODES);
  gemm1(s, apost, 13 * FDIM, P.post.w, P.post.b, t1, N_NODES, FDIM, 0);
  gemm1(s, t1, FDIM, P.lin.w, P.lin.b, t2, N_NODES, FDIM, 0);
  add_k<<<cdiv(N_NODES * FDIM, 256), 256, 0, s>>>(h, t2, N_NODES * FDIM);
}

static void run_tr(hipStream_t s, const TrP& P, float* h, float* hgn, const int* batch,
                   const int* src, const int* dst, const float* efeat,
                   float* qb, float* kb, float* vb, float* eet, float* attout,
                   float* alphab, unsigned* amaxb, float* asumb, float* t2,
                   float* macc, float* vacc, float* gcnt) {
  run_graph_norm(s, h, batch, P.gn, hgn, macc, vacc, gcnt);
  gemm1(s, hgn, FDIM, P.q.w, P.q.b, qb, N_NODES, HC, 0);
  gemm1(s, hgn, FDIM, P.k.w, P.k.b, kb, N_NODES, HC, 0);
  gemm1(s, hgn, FDIM, P.v.w, P.v.b, vb, N_NODES, HC, 0);
  gemm1(s, efeat, FEDIM, P.ew, nullptr, eet, N_EDGES, HC, 0);
  gemm1(s, hgn, FDIM, P.skip.w, P.skip.b, t2, N_NODES, FDIM, 0);
  hipMemsetAsync(amaxb, 0, (size_t)N_NODES * HEADS * 4, s);
  hipMemsetAsync(asumb, 0, (size_t)N_NODES * HEADS * 4, s);
  hipMemsetAsync(attout, 0, (size_t)N_NODES * HC * 4, s);
  attn_alpha<<<cdiv((long)N_EDGES * HEADS, 8), 256, 0, s>>>(qb, kb, eet, src, dst, alphab, amaxb, N_EDGES);
  amax_fin<<<cdiv(N_NODES * HEADS, 256), 256, 0, s>>>(amaxb, N_NODES * HEADS);
  attn_ea<<<cdiv(N_EDGES * HEADS, 256), 256, 0, s>>>(alphab, (const float*)amaxb, dst, asumb, N_EDGES * HEADS);
  attn_out<<<cdiv((long)N_EDGES * HC, 256), 256, 0, s>>>(vb, eet, alphab, asumb, src, dst, attout, N_EDGES * HC);
  trans_combine<<<cdiv(N_NODES, 8), 256, 0, s>>>(h, attout, t2, P.betaw, N_NODES);
}

// sorted-leaf sizes, for the single-concatenated-params fallback
static const int kSortedSizes[64] = {
  64, 640, 64, 4096, 64, 64, 128, 1152, 128, 128,
  64, 8192, 7, 448, 128, 128,
  128, 8192, 128, 128, 128, 128, 16384, 128, 212992, 128, 49152,
  128, 8192, 128, 128, 128, 128, 16384, 128, 212992, 128, 49152,
  384, 32768, 128, 128, 128, 512, 65536, 512, 65536, 128, 16384, 512, 65536,
  384, 32768, 128, 128, 128, 512, 65536, 512, 65536, 128, 16384, 512, 65536
};

extern "C" void kernel_launch(void* const* d_in, const int* in_sizes, int n_in,
                              void* d_out, int out_size, void* d_ws, size_t ws_size,
                              hipStream_t stream) {
  (void)out_size; (void)ws_size;
  // ---- locate inputs / params (robust to flattening convention) ----
  const float* x; const int* eidx; const float* eattr; const int* batch;
  const float* pptr[64];
  bool nestedSorted = true;
  if (in_sizes[0] == 20000) {                 // fully tree-sorted top level
    batch = (const int*)d_in[0];
    eattr = (const float*)d_in[1];
    eidx  = (const int*)d_in[2];
    for (int i = 0; i < 64; ++i) pptr[i] = (const float*)d_in[3 + i];
    x = (const float*)d_in[67];
  } else {                                    // setup_inputs() insertion order
    x     = (const float*)d_in[0];
    eidx  = (const int*)d_in[1];
    eattr = (const float*)d_in[2];
    batch = (const int*)d_in[3];
    if (n_in == 5) {                          // params concatenated in one buffer
      const float* base = (const float*)d_in[4];
      size_t off = 0;
      for (int i = 0; i < 64; ++i) { pptr[i] = base + off; off += kSortedSizes[i]; }
      nestedSorted = true;
    } else {
      for (int i = 0; i < 64 && 4 + i < n_in; ++i) pptr[i] = (const float*)d_in[4 + i];
      nestedSorted = (in_sizes[4] == 64);     // sorted order starts with edge1.b (64)
    }
  }
  AllP P;
  if (nestedSorted) fill_sorted(pptr, P); else fill_insertion(pptr, P);

  const int* src = eidx;                // edge_index[0]
  const int* dst = eidx + N_EDGES;      // edge_index[1]

  // ---- workspace carve-up ----
  float* ws = (float*)d_ws;
  size_t off = 0;
  auto take = [&](size_t n) { float* p = ws + off; off += n; return p; };
  float* h    = take((size_t)N_NODES * FDIM);
  float* hgn  = take((size_t)N_NODES * FDIM);
  float* t1   = take((size_t)N_NODES * FDIM);
  float* t2   = take((size_t)N_NODES * FDIM);
  float* sum  = take((size_t)N_NODES * FDIM);
  float* sum2 = take((size_t)N_NODES * FDIM);
  unsigned* mnb = (unsigned*)take((size_t)N_NODES * FDIM);
  unsigned* mxb = (unsigned*)take((size_t)N_NODES * FDIM);
  float* cnt  = take(N_NODES);
  float* efeat = take((size_t)N_EDGES * FEDIM);
  float* qb   = take((size_t)N_NODES * HC);
  float* kb   = take((size_t)N_NODES * HC);
  float* vb   = take((size_t)N_NODES * HC);
  float* attout = take((size_t)N_NODES * HC);
  float* alphab = take((size_t)N_EDGES * HEADS);
  unsigned* amaxb = (unsigned*)take((size_t)N_NODES * HEADS);
  float* asumb  = take((size_t)N_NODES * HEADS);
  float* macc = take(NGRAPH * FDIM);
  float* vacc = take(NGRAPH * FDIM);
  float* gcnt = take(NGRAPH);
  float* big  = take((size_t)N_EDGES * HC);   // reused region (largest: edge KV)
  // aliases within big (sequential lifetimes):
  float* e1tmp = big;
  float* e2tmp = big + (size_t)N_EDGES * FEDIM;
  float* eeenc = big;
  float* medge = big + (size_t)N_EDGES * FDIM;
  float* apost = big + (size_t)N_EDGES * FDIM * 2;
  float* eet   = big;

  hipStream_t s = stream;

  // ---- node embedding ----
  gemm1(s, x, 9, P.emb.w, P.emb.b, t1, N_NODES, FDIM, 1);
  layernorm_k<<<cdiv(N_NODES, 8), 256, 0, s>>>(t1, h, P.emb_ln_g, P.emb_ln_b, N_NODES, FDIM);
  // ---- edge MLP ----
  gemm1(s, eattr, 10, P.edge1.w, P.edge1.b, e1tmp, N_EDGES, FEDIM, 1);
  gemm1(s, e1tmp, FEDIM, P.edge2.w, P.edge2.b, e2tmp, N_EDGES, FEDIM, 1);
  layernorm_k<<<cdiv(N_EDGES, 8), 256, 0, s>>>(e2tmp, efeat, P.edge_ln_g, P.edge_ln_b, N_EDGES, FEDIM);

  // ---- 4 GNN layers ----
  run_pna(s, P.pna1, h, hgn, batch, src, dst, efeat, sum, sum2, mnb, mxb, cnt,
          eeenc, medge, apost, t1, t2, macc, vacc, gcnt);
  run_tr(s, P.tr1, h, hgn, batch, src, dst, efeat, qb, kb, vb, eet, attout,
         alphab, amaxb, asumb, t2, macc, vacc, gcnt);
  run_pna(s, P.pna2, h, hgn, batch, src, dst, efeat, sum, sum2, mnb, mxb, cnt,
          eeenc, medge, apost, t1, t2, macc, vacc, gcnt);
  run_tr(s, P.tr2, h, hgn, batch, src, dst, efeat, qb, kb, vb, eet, attout,
         alphab, amaxb, asumb, t2, macc, vacc, gcnt);

  // ---- head ----
  layernorm_k<<<cdiv(N_NODES, 8), 256, 0, s>>>(h, hgn, P.head_ln_g, P.head_ln_b, N_NODES, FDIM);
  gemm1(s, hgn, FDIM, P.head1.w, P.head1.b, t1, N_NODES, 64, 1);
  gemm1(s, t1, 64, P.head2.w, P.head2.b, (float*)d_out, N_NODES, 7, 0);
}
